// NeighbourMultiHeadAttention_49778670960966
// MI455X (gfx1250) — matrix operand order repaired and checked by reading, not compiled, not run on hardware
//
#include <hip/hip_runtime.h>
#include <hip/hip_bf16.h>

typedef __attribute__((ext_vector_type(16))) _Float16 v16h;
typedef __attribute__((ext_vector_type(8)))  _Float16 v8h;
typedef __attribute__((ext_vector_type(4)))  _Float16 v4h;
typedef __attribute__((ext_vector_type(8)))  float    v8f;

#define WMMA_F16(A, B, C) \
    __builtin_amdgcn_wmma_f32_16x16x32_f16(false, (A), false, (B), (short)0, (C), false, false)

#define CAT16(lo, hi) \
    __builtin_shufflevector((lo), (hi), 0, 1, 2, 3, 4, 5, 6, 7, 8, 9, 10, 11, 12, 13, 14, 15)

// Weight LDS stride (halves): padded to keep 16B alignment and spread banks.
#define WSTRIDE 72

__device__ __forceinline__ unsigned floatToKey(float f) {
    unsigned u = __float_as_uint(f);
    return (u & 0x80000000u) ? ~u : (u | 0x80000000u);
}
__device__ __forceinline__ float keyToFloat(unsigned k) {
    unsigned u = (k & 0x80000000u) ? (k ^ 0x80000000u) : ~k;
    return __uint_as_float(u);
}

// Sum with lane^8 partner via immediate ds_swizzle (SWAPX8: xor=8, and=0x1f).
__device__ __forceinline__ float xor8_add(float p) {
    float q = __int_as_float(__builtin_amdgcn_ds_swizzle(__float_as_int(p), 0x201F));
    return p + q;
}

// A operand (16x32 f16). Lane layout (ISA 7.12.2): lanes 0-15 hold K 8*hi+{0..7}
// and +16; both runs contiguous in a row-major f16 row -> two b128 LDS loads.
__device__ __forceinline__ v16h loadA(const _Float16* rowPtr, int kc, int hi) {
    const v8h* p = (const v8h*)(rowPtr + kc * 32 + 8 * hi);
    v8h lo = p[0];
    v8h hv = p[2];   // +16 halves
    return CAT16(lo, hv);
}

// B operand (32x16 f16) from transposed weight LDS (wT[n*WSTRIDE + k]):
// lane needs k = 32*kc + 8*hi + {0..7} and +16 (same as A) -> two b128 LDS loads.
__device__ __forceinline__ v16h loadB(const _Float16* colPtr, int kc, int hi) {
    const v8h* p = (const v8h*)(colPtr + kc * 32 + 8 * hi);
    v8h lo = p[0];
    v8h hv = p[2];   // +16 halves
    return CAT16(lo, hv);
}

__device__ __forceinline__ v4h cvt4(float4 a) {
    v4h r;
    r[0] = (_Float16)a.x; r[1] = (_Float16)a.y;
    r[2] = (_Float16)a.z; r[3] = (_Float16)a.w;
    return r;
}

// ---------------- init: maxKey = key(-inf), nodeSum = 0, agg = 0 ----------------
__global__ __launch_bounds__(256) void k0_init(unsigned* __restrict__ maxKey,
                                               float* __restrict__ nodeSum,
                                               float* __restrict__ agg, int N) {
    long long i = (long long)blockIdx.x * 256 + threadIdx.x;
    long long nh = (long long)N * 8;
    if (i < nh) { maxKey[i] = 0x007FFFFFu; nodeSum[i] = 0.0f; }
    long long na = (long long)N * 64;
    if (i < na) agg[i] = 0.0f;
}

// ---------------- K1: logits[E,8] = per-head q.k / sqrt(8); segment max ----------------
__global__ __launch_bounds__(256) void k1_logits(
        const float* __restrict__ own, const float* __restrict__ srcm,
        const int* __restrict__ idx,
        const float* __restrict__ Wq, const float* __restrict__ bq,
        const float* __restrict__ Wk, const float* __restrict__ bk,
        float* __restrict__ logits, unsigned* __restrict__ maxKey, int E) {
    __shared__ _Float16 ownH[128 * 64];
    __shared__ _Float16 srcH[128 * 64];
    __shared__ _Float16 wqT[64 * WSTRIDE];
    __shared__ _Float16 wkT[64 * WSTRIDE];
    __shared__ float bqS[64], bkS[64];

    const int tid = threadIdx.x;
    const long long e0 = (long long)blockIdx.x * 128;

    // weights: read row-major (coalesced), store transposed [n][k] as f16
    for (int i = tid; i < 4096; i += 256) {
        int k = i >> 6, n = i & 63;
        wqT[n * WSTRIDE + k] = (_Float16)Wq[i];
        wkT[n * WSTRIDE + k] = (_Float16)Wk[i];
    }
    if (tid < 64) { bqS[tid] = bq[tid]; bkS[tid] = bk[tid]; }

    // activations: coalesced float4 loads, converted to f16 rows in LDS
    const float4* og = (const float4*)(own + e0 * 64);
    const float4* sg = (const float4*)(srcm + e0 * 64);
    v4h* o4 = (v4h*)ownH;
    v4h* s4 = (v4h*)srcH;
    for (int i = tid; i < 2048; i += 256) {
        long long row = e0 + (i >> 4);
        float4 z = {0.f, 0.f, 0.f, 0.f};
        float4 o = (row < E) ? og[i] : z;
        float4 s = (row < E) ? sg[i] : z;
        o4[i] = cvt4(o);
        s4[i] = cvt4(s);
    }
    __syncthreads();

    const int lane = tid & 31;
    const int wave = tid >> 5;
    const int eL = wave * 16;
    const int m  = lane & 15;
    const int hi = lane >> 4;

    const _Float16* orow = &ownH[(eL + m) * 64];
    const _Float16* srow = &srcH[(eL + m) * 64];
    v16h aq0 = loadA(orow, 0, hi);
    v16h aq1 = loadA(orow, 1, hi);
    v16h ak0 = loadA(srow, 0, hi);
    v16h ak1 = loadA(srow, 1, hi);

    float acc[8];
#pragma unroll
    for (int r = 0; r < 8; ++r) acc[r] = 0.0f;

#pragma unroll
    for (int t = 0; t < 4; ++t) {
        const int n = t * 16 + (lane & 15);
        const _Float16* qcol = &wqT[n * WSTRIDE];
        const _Float16* kcol = &wkT[n * WSTRIDE];
        v16h bqf0 = loadB(qcol, 0, hi);
        v16h bqf1 = loadB(qcol, 1, hi);
        v16h bkf0 = loadB(kcol, 0, hi);
        v16h bkf1 = loadB(kcol, 1, hi);

        v8f cq = {};
        cq = WMMA_F16(aq0, bqf0, cq);
        cq = WMMA_F16(aq1, bqf1, cq);
        v8f ck = {};
        ck = WMMA_F16(ak0, bkf0, ck);
        ck = WMMA_F16(ak1, bkf1, ck);

        const float qb = bqS[n];
        const float kb = bkS[n];
#pragma unroll
        for (int r = 0; r < 8; ++r) {
            float p = (cq[r] + qb) * (ck[r] + kb);
            acc[r] += xor8_add(p);     // fold the two a-values sharing a head
        }
    }

    if ((lane & 8) == 0) {             // lanes 0-7 (edges r), 16-23 (edges r+8)
        const float s = 0.35355339059327373f;  // 1/sqrt(8)
        const int h = lane & 7;
#pragma unroll
        for (int r = 0; r < 8; ++r) {
            long long e = e0 + eL + r + 8 * hi;
            if (e < E) {
                float lg = acc[r] * s;
                logits[e * 8 + h] = lg;
                int node = idx[e];
                atomicMax(&maxKey[(long long)node * 8 + h], floatToKey(lg));
            }
        }
    }
}

// ---------------- K2: ex = exp(logit - segmax); segment sum ----------------
__global__ __launch_bounds__(256) void k2_exp(
        const int* __restrict__ idx, float* __restrict__ exbuf,
        const unsigned* __restrict__ maxKey, float* __restrict__ nodeSum, long long EH) {
    long long g = (long long)blockIdx.x * 256 + threadIdx.x;
    if (g >= EH) return;
    long long e = g >> 3;
    int h = (int)(g & 7);
    int node = idx[e];
    float mk = keyToFloat(maxKey[(long long)node * 8 + h]);
    float ex = expf(exbuf[g] - mk);
    exbuf[g] = ex;
    atomicAdd(&nodeSum[(long long)node * 8 + h], ex);
}

// ---------------- K3: V projection, attn weight, scatter-add into agg[N,64] ----------------
__global__ __launch_bounds__(256) void k3_value(
        const float* __restrict__ srcm, const int* __restrict__ idx,
        const float* __restrict__ Wv, const float* __restrict__ bv,
        const float* __restrict__ exbuf, const float* __restrict__ nodeSum,
        float* __restrict__ agg, int E) {
    __shared__ _Float16 srcH[128 * 64];
    __shared__ _Float16 wvT[64 * WSTRIDE];
    __shared__ float bvS[64];
    __shared__ float attnS[128 * 8];
    __shared__ int idxS[128];

    const int tid = threadIdx.x;
    const long long e0 = (long long)blockIdx.x * 128;

    for (int i = tid; i < 4096; i += 256) {
        int k = i >> 6, n = i & 63;
        wvT[n * WSTRIDE + k] = (_Float16)Wv[i];
    }
    if (tid < 64) bvS[tid] = bv[tid];

    const float4* sg = (const float4*)(srcm + e0 * 64);
    v4h* s4 = (v4h*)srcH;
    for (int i = tid; i < 2048; i += 256) {
        long long row = e0 + (i >> 4);
        float4 z = {0.f, 0.f, 0.f, 0.f};
        float4 s = (row < E) ? sg[i] : z;
        s4[i] = cvt4(s);
    }
    for (int i = tid; i < 128; i += 256) idxS[i] = (e0 + i < E) ? idx[e0 + i] : 0;
    __syncthreads();

    for (int i = tid; i < 1024; i += 256) {
        int el = i >> 3, h = i & 7;
        long long e = e0 + el;
        float a = 0.0f;
        if (e < E) {
            int node = idxS[el];
            a = exbuf[e * 8 + h] / (nodeSum[(long long)node * 8 + h] + 1e-16f);
        }
        attnS[i] = a;
    }
    __syncthreads();

    const int lane = tid & 31;
    const int wave = tid >> 5;
    const int eL = wave * 16;
    const int m  = lane & 15;
    const int hi = lane >> 4;

    const _Float16* srow = &srcH[(eL + m) * 64];
    v16h av0 = loadA(srow, 0, hi);
    v16h av1 = loadA(srow, 1, hi);

#pragma unroll
    for (int t = 0; t < 4; ++t) {
        const int n = t * 16 + (lane & 15);
        const _Float16* vcol = &wvT[n * WSTRIDE];
        v16h bvf0 = loadB(vcol, 0, hi);
        v16h bvf1 = loadB(vcol, 1, hi);

        v8f cv = {};
        cv = WMMA_F16(av0, bvf0, cv);
        cv = WMMA_F16(av1, bvf1, cv);

        const float vb = bvS[n];
        const int h = lane & 7;        // head = column % 8
#pragma unroll
        for (int r = 0; r < 8; ++r) {
            int el = eL + r + 8 * hi;
            long long e = e0 + el;
            if (e < E) {
                float val = (cv[r] + vb) * attnS[el * 8 + h];
                atomicAdd(&agg[(long long)idxS[el] * 64 + n], val);
            }
        }
    }
}

// ---------------- K4: out = agg @ Wo + bo ----------------
__global__ __launch_bounds__(256) void k4_out(
        const float* __restrict__ agg, const float* __restrict__ Wo,
        const float* __restrict__ bo, float* __restrict__ out, int N) {
    __shared__ _Float16 aggH[128 * 64];
    __shared__ _Float16 woT[64 * WSTRIDE];
    __shared__ float boS[64];

    const int tid = threadIdx.x;
    const long long r0 = (long long)blockIdx.x * 128;

    for (int i = tid; i < 4096; i += 256) {
        int k = i >> 6, n = i & 63;
        woT[n * WSTRIDE + k] = (_Float16)Wo[i];
    }
    if (tid < 64) boS[tid] = bo[tid];

    const float4* ag = (const float4*)(agg + r0 * 64);
    v4h* a4 = (v4h*)aggH;
    for (int i = tid; i < 2048; i += 256) {
        long long row = r0 + (i >> 4);
        float4 z = {0.f, 0.f, 0.f, 0.f};
        float4 a = (row < N) ? ag[i] : z;
        a4[i] = cvt4(a);
    }
    __syncthreads();

    const int lane = tid & 31;
    const int wave = tid >> 5;
    const int rL = wave * 16;
    const int m  = lane & 15;
    const int hi = lane >> 4;

    const _Float16* arow = &aggH[(rL + m) * 64];
    v16h aa0 = loadA(arow, 0, hi);
    v16h aa1 = loadA(arow, 1, hi);

#pragma unroll
    for (int t = 0; t < 4; ++t) {
        const int n = t * 16 + (lane & 15);
        const _Float16* ocol = &woT[n * WSTRIDE];
        v16h bof0 = loadB(ocol, 0, hi);
        v16h bof1 = loadB(ocol, 1, hi);

        v8f c = {};
        c = WMMA_F16(aa0, bof0, c);
        c = WMMA_F16(aa1, bof1, c);
#pragma unroll
        for (int r = 0; r < 8; ++r) {
            long long row = r0 + rL + r + 8 * hi;
            if (row < N) out[row * 64 + n] = c[r] + boS[n];
        }
    }
}

extern "C" void kernel_launch(void* const* d_in, const int* in_sizes, int n_in,
                              void* d_out, int out_size, void* d_ws, size_t ws_size,
                              hipStream_t stream) {
    const float* own  = (const float*)d_in[0];
    const float* srcm = (const float*)d_in[1];
    const int*   idx  = (const int*)d_in[2];
    const float* Wq = (const float*)d_in[3];
    const float* bq = (const float*)d_in[4];
    const float* Wk = (const float*)d_in[5];
    const float* bk = (const float*)d_in[6];
    const float* Wv = (const float*)d_in[7];
    const float* bv = (const float*)d_in[8];
    const float* Wo = (const float*)d_in[9];
    const float* bo = (const float*)d_in[10];
    float* out = (float*)d_out;

    const int E = in_sizes[0] / 64;   // edges
    const int N = out_size / 64;      // nodes

    // workspace layout (f32/u32, all 4B aligned):
    //   exbuf  [E*8]   logits, overwritten with exp()
    //   maxKey [N*8]   order-preserving uint keys for segment max
    //   nodeSum[N*8]   segment softmax denominators
    //   agg    [N*64]  attention-weighted value accumulation
    float*    exbuf   = (float*)d_ws;
    unsigned* maxKey  = (unsigned*)(exbuf + (size_t)E * 8);
    float*    nodeSum = (float*)(maxKey + (size_t)N * 8);
    float*    agg     = nodeSum + (size_t)N * 8;

    {
        long long n = (long long)N * 64;
        int blocks = (int)((n + 255) / 256);
        k0_init<<<blocks, 256, 0, stream>>>(maxKey, nodeSum, agg, N);
    }
    {
        int blocks = (E + 127) / 128;
        k1_logits<<<blocks, 256, 0, stream>>>(own, srcm, idx, Wq, bq, Wk, bk,
                                              exbuf, maxKey, E);
    }
    {
        long long eh = (long long)E * 8;
        int blocks = (int)((eh + 255) / 256);
        k2_exp<<<blocks, 256, 0, stream>>>(idx, exbuf, maxKey, nodeSum, eh);
    }
    {
        int blocks = (E + 127) / 128;
        k3_value<<<blocks, 256, 0, stream>>>(srcm, idx, Wv, bv, exbuf, nodeSum, agg, E);
    }
    {
        int tiles = (N + 15) / 16;
        int blocks = (tiles + 7) / 8;
        k4_out<<<blocks, 256, 0, stream>>>(agg, Wo, bo, out, N);
    }
}
